// MultiheadLatentAttentionV3_2_59897613910166
// MI455X (gfx1250) — compile-verified, matrix operand order checked by reading, not run on hardware
//
#include <hip/hip_runtime.h>
#include <hip/hip_bf16.h>

// ---------------------------------------------------------------------------
// MLA forward for gfx1250 (MI455X): all dense matmuls + attention run through
// v_wmma_f32_16x16x32_bf16 (wave32 WMMA), f32 accumulation.
// GEMMs use 128x128 tiles with double-buffered LDS filled via the CDNA5 async
// copy path (global_load_async_to_lds_b128 + s_wait_asynccnt).
// ---------------------------------------------------------------------------

typedef __attribute__((ext_vector_type(16))) __bf16 v16bf;
typedef __attribute__((ext_vector_type(8)))  __bf16 v8bf;
typedef __attribute__((ext_vector_type(8)))  float  v8f;

#define NH_   16
#define IH_   8
#define NOPE_ 128
#define ROPE_ 64
#define DQK_  192
#define VD_   128
#define TOPK_ 8
#define B_    2
#define S_    2048
#define H_    2048
#define QLR_  1536

__device__ inline __bf16 f2bf(float f) {
  unsigned u = __builtin_bit_cast(unsigned, f);
  u += 0x7FFFu + ((u >> 16) & 1u);                 // round to nearest even
  unsigned short h = (unsigned short)(u >> 16);
  return __builtin_bit_cast(__bf16, h);
}

__device__ inline v16bf cat16(v8bf lo, v8bf hi) {
  return __builtin_shufflevector(lo, hi, 0,1,2,3,4,5,6,7,8,9,10,11,12,13,14,15);
}

// CDNA5 async copy: 16B global -> LDS, tracked by ASYNCcnt.
// lds_off = wave-relative LDS byte address (low 32 bits of the generic address,
// per ISA 10.2 aperture rules).
__device__ inline void async_ld_b128(void* lds_ptr, const void* gptr) {
  unsigned off = (unsigned)(size_t)lds_ptr;
  asm volatile("global_load_async_to_lds_b128 %0, %1, off"
               :: "v"(off), "v"(gptr) : "memory");
}
__device__ inline void wait_async_le4() { asm volatile("s_wait_asynccnt 4" ::: "memory"); }
__device__ inline void wait_async_0()   { asm volatile("s_wait_asynccnt 0" ::: "memory"); }

// ---------------------------------------------------------------------------
// Elementwise converts
// ---------------------------------------------------------------------------
__global__ void cvt_f32_bf16_kernel(__bf16* __restrict__ dst, const float* __restrict__ src, size_t n) {
  size_t i = (size_t)blockIdx.x * blockDim.x + threadIdx.x;
  if (i < n) dst[i] = f2bf(src[i]);
}

// dst[n*K + k] = bf16(src[k*N + n])  (weight transpose for B^T operand)
__global__ void wt_transpose_kernel(__bf16* __restrict__ dst, const float* __restrict__ src, int K, int N) {
  size_t i = (size_t)blockIdx.x * blockDim.x + threadIdx.x;
  if (i >= (size_t)K * N) return;
  int n = (int)(i / K), k = (int)(i % K);
  dst[i] = f2bf(src[(size_t)k * N + n]);
}

// ---------------------------------------------------------------------------
// WMMA GEMM: C[M,N](f32) = A[M,K](bf16, row-major) * BT[N,K](bf16)^T
// Tile 128x128, 256 threads = 8 waves (4x2), each wave 32x64 (2x4 WMMA frags),
// K step 32, double-buffered LDS via async copies.
// ---------------------------------------------------------------------------
#define TM 128
#define TN 128
#define TK 32

__global__ __launch_bounds__(256)
void gemm_bf16_wmma_kernel(float* __restrict__ C, const __bf16* __restrict__ A,
                           const __bf16* __restrict__ BT, int M, int N, int K) {
  __shared__ alignas(16) __bf16 As[2][TM][TK + 8];   // 2*128*40*2B = 20KB
  __shared__ alignas(16) __bf16 Bs[2][TN][TK + 8];   // 20KB
  int tid  = threadIdx.x;
  int wave = tid >> 5;
  int lane = tid & 31;
  int n16  = lane & 15;
  int hi8  = (lane >> 4) * 8;
  int m0 = blockIdx.y * TM;
  int n0 = blockIdx.x * TN;
  int wm = (wave & 3) * 32;       // 4 waves along M
  int wn = (wave >> 2) * 64;      // 2 waves along N

  int lr = tid >> 1;              // 0..127
  int lc = (tid & 1) * 16;        // 0 or 16

  v8f acc[2][4] = {};

  // issue tile 0 (4 async B128 per thread)
  {
    const __bf16* sA = A  + (size_t)(m0 + lr) * K + lc;
    const __bf16* sB = BT + (size_t)(n0 + lr) * K + lc;
    async_ld_b128(&As[0][lr][lc],     sA);
    async_ld_b128(&As[0][lr][lc + 8], sA + 8);
    async_ld_b128(&Bs[0][lr][lc],     sB);
    async_ld_b128(&Bs[0][lr][lc + 8], sB + 8);
  }

  int buf = 0;
  for (int kk = 0; kk < K; kk += TK) {
    bool more = (kk + TK) < K;
    if (more) {  // stream next tile into the other buffer while computing
      const __bf16* sA = A  + (size_t)(m0 + lr) * K + kk + TK + lc;
      const __bf16* sB = BT + (size_t)(n0 + lr) * K + kk + TK + lc;
      async_ld_b128(&As[buf ^ 1][lr][lc],     sA);
      async_ld_b128(&As[buf ^ 1][lr][lc + 8], sA + 8);
      async_ld_b128(&Bs[buf ^ 1][lr][lc],     sB);
      async_ld_b128(&Bs[buf ^ 1][lr][lc + 8], sB + 8);
      wait_async_le4();   // async loads complete in order: current tile landed
    } else {
      wait_async_0();
    }
    __syncthreads();

    v16bf af[2], bfr[4];
#pragma unroll
    for (int mi = 0; mi < 2; ++mi) {
      int row = wm + mi * 16 + n16;
      af[mi] = cat16(*(const v8bf*)&As[buf][row][hi8], *(const v8bf*)&As[buf][row][16 + hi8]);
    }
#pragma unroll
    for (int ni = 0; ni < 4; ++ni) {
      int col = wn + ni * 16 + n16;
      bfr[ni] = cat16(*(const v8bf*)&Bs[buf][col][hi8], *(const v8bf*)&Bs[buf][col][16 + hi8]);
    }
#pragma unroll
    for (int mi = 0; mi < 2; ++mi)
#pragma unroll
      for (int ni = 0; ni < 4; ++ni)
        acc[mi][ni] = __builtin_amdgcn_wmma_f32_16x16x32_bf16(
            false, af[mi], false, bfr[ni], (short)0, acc[mi][ni], false, false);
    __syncthreads();    // all waves done reading buf before it is overwritten
    buf ^= 1;
  }

#pragma unroll
  for (int mi = 0; mi < 2; ++mi)
#pragma unroll
    for (int ni = 0; ni < 4; ++ni)
#pragma unroll
      for (int rr = 0; rr < 8; ++rr) {
        int m = m0 + wm + mi * 16 + hi8 + rr;
        int n = n0 + wn + ni * 16 + n16;
        C[(size_t)m * N + n] = acc[mi][ni][rr];
      }
}

// ---------------------------------------------------------------------------
// LayerNorm over rows of length N (qc path), writes bf16.
// ---------------------------------------------------------------------------
__global__ void ln_rows_kernel(__bf16* __restrict__ out, const float* __restrict__ in,
                               const float* __restrict__ w, const float* __restrict__ b, int N) {
  int row = blockIdx.x;
  const float* r = in + (size_t)row * N;
  __shared__ float red[256];
  float s = 0.f;
  for (int i = threadIdx.x; i < N; i += 256) s += r[i];
  red[threadIdx.x] = s; __syncthreads();
  for (int st = 128; st > 0; st >>= 1) { if (threadIdx.x < st) red[threadIdx.x] += red[threadIdx.x + st]; __syncthreads(); }
  float mu = red[0] / (float)N; __syncthreads();
  float v = 0.f;
  for (int i = threadIdx.x; i < N; i += 256) { float d = r[i] - mu; v += d * d; }
  red[threadIdx.x] = v; __syncthreads();
  for (int st = 128; st > 0; st >>= 1) { if (threadIdx.x < st) red[threadIdx.x] += red[threadIdx.x + st]; __syncthreads(); }
  float rstd = rsqrtf(red[0] / (float)N + 1e-5f); __syncthreads();
  for (int i = threadIdx.x; i < N; i += 256)
    out[(size_t)row * N + i] = f2bf((r[i] - mu) * rstd * w[i] + b[i]);
}

// kv layernorm: per (b,h,s), LN over 128 dims taken from kvc cols [h*192+64, h*192+192)
__global__ void kv_ln_kernel(__bf16* __restrict__ out, const float* __restrict__ kvc,
                             const float* __restrict__ w, const float* __restrict__ bta, int S) {
  int gid = blockIdx.x;                     // (b*16 + h)*S + s
  int s = gid % S; int h = (gid / S) & 15; int b = gid / (S * 16);
  const float* src = kvc + ((size_t)(b * S + s)) * 3072 + h * DQK_ + ROPE_;
  int t = threadIdx.x;                      // 128 threads
  float v = src[t];
  __shared__ float red[128];
  red[t] = v; __syncthreads();
  for (int st = 64; st > 0; st >>= 1) { if (t < st) red[t] += red[t + st]; __syncthreads(); }
  float mu = red[0] * (1.0f / 128.0f); __syncthreads();
  float d = v - mu;
  red[t] = d * d; __syncthreads();
  for (int st = 64; st > 0; st >>= 1) { if (t < st) red[t] += red[t + st]; __syncthreads(); }
  float rstd = rsqrtf(red[0] * (1.0f / 128.0f) + 1e-5f);
  out[(size_t)gid * 128 + t] = f2bf(d * rstd * w[t] + bta[t]);
}

// ---------------------------------------------------------------------------
// Build Q / K (with RoPE on last 64 dims) and V^T, all bf16
// ---------------------------------------------------------------------------
__device__ inline float rope_val(const float* pair_base, int r, int s) {
  int pi = r >> 1;
  float ang = (float)s * __expf(-9.2103403720f * (float)(2 * pi) * (1.0f / 64.0f));
  float sn, cs; __sincosf(ang, &sn, &cs);
  float xr = pair_base[2 * pi], xi = pair_base[2 * pi + 1];
  return (r & 1) ? (xr * sn + xi * cs) : (xr * cs - xi * sn);
}

__global__ void build_q_kernel(__bf16* __restrict__ Q, const float* __restrict__ qp, int S) {
  size_t i = (size_t)blockIdx.x * 256 + threadIdx.x;   // over B*16*S*192
  if (i >= (size_t)B_ * NH_ * S * DQK_) return;
  int d = (int)(i % DQK_); size_t t = i / DQK_;
  int s = (int)(t % S); int h = (int)((t / S) & 15); int b = (int)(t / ((size_t)S * 16));
  const float* row = qp + ((size_t)(b * S + s)) * 3072 + h * DQK_;
  float val = (d < NOPE_) ? row[d] : rope_val(row + NOPE_, d - NOPE_, s);
  Q[i] = f2bf(val);
}

__global__ void build_k_kernel(__bf16* __restrict__ K, const float* __restrict__ kvp,
                               const float* __restrict__ kvc, int S) {
  size_t i = (size_t)blockIdx.x * 256 + threadIdx.x;
  if (i >= (size_t)B_ * NH_ * S * DQK_) return;
  int d = (int)(i % DQK_); size_t t = i / DQK_;
  int s = (int)(t % S); int h = (int)((t / S) & 15); int b = (int)(t / ((size_t)S * 16));
  float val;
  if (d < NOPE_) {
    val = kvp[t * 256 + d];                            // k_nope
  } else {
    const float* row = kvc + ((size_t)(b * S + s)) * 3072 + h * DQK_;  // k_pe = first 64
    val = rope_val(row, d - NOPE_, s);
  }
  K[i] = f2bf(val);
}

__global__ void build_vt_kernel(__bf16* __restrict__ VT, const float* __restrict__ kvp, int S) {
  size_t i = (size_t)blockIdx.x * 256 + threadIdx.x;   // over B*16*128*S
  if (i >= (size_t)B_ * NH_ * VD_ * S) return;
  int s = (int)(i % S); int d = (int)((i / S) % VD_); int bh = (int)(i / ((size_t)S * VD_));
  VT[i] = f2bf(kvp[((size_t)bh * S + s) * 256 + 128 + d]);
}

// ---------------------------------------------------------------------------
// Flash attention, causal, d=192, v=128.  grid=(S/64, B*NH), block=128 (4 waves)
// Each wave owns 16 q rows; K tile of 32 per iteration; WMMA for QK^T and P*V.
// ---------------------------------------------------------------------------
#define FA_SCALE 0.07216878364870322f   // 1/sqrt(192)

__global__ __launch_bounds__(128)
void flash_attn_kernel(float* __restrict__ O, const __bf16* __restrict__ Q,
                       const __bf16* __restrict__ Km, const __bf16* __restrict__ VT, int S) {
  int qt   = blockIdx.x;
  int bh   = blockIdx.y;
  int wave = threadIdx.x >> 5;
  int lane = threadIdx.x & 31;
  int n16  = lane & 15;
  int hi8  = (lane >> 4) * 8;
  int qrb  = qt * 64 + wave * 16;

  __shared__ alignas(16) __bf16 Pbuf[4][16][40];

  // preload Q fragments (A-layout, 6 chunks of K=32 covering d=192)
  const __bf16* Qrow = Q + ((size_t)bh * S + (qrb + n16)) * DQK_;
  v16bf qf[6];
#pragma unroll
  for (int c = 0; c < 6; ++c)
    qf[c] = cat16(*(const v8bf*)(Qrow + c * 32 + hi8),
                  *(const v8bf*)(Qrow + c * 32 + 16 + hi8));

  v8f o[8] = {};
  float rmax[8], rsum[8];
#pragma unroll
  for (int r = 0; r < 8; ++r) { rmax[r] = -1e30f; rsum[r] = 0.0f; }

  int nkt = 2 * qt + 2;
  for (int kt = 0; kt < nkt; ++kt) {
    int kb = kt * 32;
    // prefetch K rows two tiles ahead (emits global_prefetch_b8)
    if (kb + 64 + n16 < S)
      __builtin_prefetch(Km + ((size_t)bh * S + (kb + 64 + n16)) * DQK_, 0, 3);

    v8f sc[2];
#pragma unroll
    for (int nt = 0; nt < 2; ++nt) {
      const __bf16* Krow = Km + ((size_t)bh * S + (kb + nt * 16 + n16)) * DQK_;
      v8f a = {};
#pragma unroll
      for (int c = 0; c < 6; ++c) {
        v16bf kf = cat16(*(const v8bf*)(Krow + c * 32 + hi8),
                         *(const v8bf*)(Krow + c * 32 + 16 + hi8));
        a = __builtin_amdgcn_wmma_f32_16x16x32_bf16(false, qf[c], false, kf, (short)0, a, false, false);
      }
      int ncol = kb + nt * 16 + n16;
#pragma unroll
      for (int r = 0; r < 8; ++r) {
        float v = a[r] * FA_SCALE;
        int qrow = qrb + hi8 + r;
        a[r] = (ncol > qrow) ? -1e30f : v;
      }
      sc[nt] = a;
    }
    // online softmax (per-row stats replicated across each 16-lane half)
    float alpha[8];
#pragma unroll
    for (int r = 0; r < 8; ++r) {
      float tm = fmaxf(sc[0][r], sc[1][r]);
#pragma unroll
      for (int m = 1; m <= 8; m <<= 1) tm = fmaxf(tm, __shfl_xor(tm, m, 32));
      float nm = fmaxf(rmax[r], tm);
      alpha[r] = __expf(rmax[r] - nm);
      rmax[r] = nm;
      float p0 = __expf(sc[0][r] - nm);
      float p1 = __expf(sc[1][r] - nm);
      float ps = p0 + p1;
#pragma unroll
      for (int m = 1; m <= 8; m <<= 1) ps += __shfl_xor(ps, m, 32);
      rsum[r] = rsum[r] * alpha[r] + ps;
      Pbuf[wave][hi8 + r][n16]      = f2bf(p0);
      Pbuf[wave][hi8 + r][16 + n16] = f2bf(p1);
    }
#pragma unroll
    for (int vt = 0; vt < 8; ++vt)
#pragma unroll
      for (int r = 0; r < 8; ++r) o[vt][r] *= alpha[r];
    __syncthreads();

    const __bf16* prow = &Pbuf[wave][n16][0];
    v16bf pa = cat16(*(const v8bf*)(prow + hi8), *(const v8bf*)(prow + 16 + hi8));
#pragma unroll
    for (int vt = 0; vt < 8; ++vt) {
      const __bf16* Vrow = VT + ((size_t)bh * VD_ + vt * 16 + n16) * S + kb;
      v16bf vf = cat16(*(const v8bf*)(Vrow + hi8), *(const v8bf*)(Vrow + 16 + hi8));
      o[vt] = __builtin_amdgcn_wmma_f32_16x16x32_bf16(false, pa, false, vf, (short)0, o[vt], false, false);
    }
    __syncthreads();
  }

#pragma unroll
  for (int vt = 0; vt < 8; ++vt)
#pragma unroll
    for (int r = 0; r < 8; ++r) {
      int qrow = qrb + hi8 + r;
      O[((size_t)bh * S + qrow) * VD_ + vt * 16 + n16] = o[vt][r] / rsum[r];
    }
}

// ---------------------------------------------------------------------------
// Sparse side branch
// ---------------------------------------------------------------------------
__global__ void gate_kernel(float* __restrict__ gate, const float* __restrict__ x,
                            const float* __restrict__ Wig, int rows) {
  int o = blockIdx.x * 256 + threadIdx.x;
  if (o >= rows * IH_) return;
  int ih = o % IH_; int row = o / IH_;
  const float* xr = x + (size_t)row * H_;
  float acc = 0.f;
  for (int k = 0; k < H_; ++k) acc += xr[k] * Wig[(size_t)k * IH_ + ih];
  gate[o] = acc;
}

__global__ void topk_kernel(int* __restrict__ topidx, const float* __restrict__ gate, int S) {
  int b = blockIdx.x / IH_; int ih = blockIdx.x % IH_;
  __shared__ float bv[256]; __shared__ int bi[256]; __shared__ int chosen[TOPK_];
  for (int t = 0; t < TOPK_; ++t) {
    float best = -3.4e38f; int besti = -1;
    for (int s = threadIdx.x; s < S; s += 256) {
      bool used = false;
      for (int j = 0; j < t; ++j) if (chosen[j] == s) used = true;
      float v = gate[((size_t)(b * S + s)) * IH_ + ih];
      if (!used && v > best) { best = v; besti = s; }
    }
    bv[threadIdx.x] = best; bi[threadIdx.x] = besti; __syncthreads();
    for (int st = 128; st > 0; st >>= 1) {
      if (threadIdx.x < st) {
        bool take = (bv[threadIdx.x + st] > bv[threadIdx.x]) ||
                    (bv[threadIdx.x + st] == bv[threadIdx.x] && bi[threadIdx.x + st] >= 0 &&
                     (bi[threadIdx.x] < 0 || bi[threadIdx.x + st] < bi[threadIdx.x]));
        if (take) { bv[threadIdx.x] = bv[threadIdx.x + st]; bi[threadIdx.x] = bi[threadIdx.x + st]; }
      }
      __syncthreads();
    }
    if (threadIdx.x == 0) { chosen[t] = bi[0]; topidx[blockIdx.x * TOPK_ + t] = bi[0]; }
    __syncthreads();
  }
}

// One block per (b, ih): gather 8 tokens, sq/sk/sv, 8x8 attention, mean, @W_io[:, :128]
__global__ void sparse_kernel(float* __restrict__ io_vec, const float* __restrict__ ip,
                              const int* __restrict__ topidx,
                              const float* __restrict__ Wsq, const float* __restrict__ Wsk,
                              const float* __restrict__ Wsv, const float* __restrict__ Wio, int S) {
  int b = blockIdx.x / IH_, ih = blockIdx.x % IH_;
  __shared__ float sel[TOPK_][128], sq[TOPK_][128], sk[TOPK_][128], sv[TOPK_][128];
  __shared__ float sp[TOPK_][TOPK_], mean[128];
  const int* idx = topidx + blockIdx.x * TOPK_;
  for (int e = threadIdx.x; e < TOPK_ * 128; e += 256) {
    int t = e >> 7, d = e & 127;
    sel[t][d] = ip[((size_t)(b * S + idx[t])) * 1024 + ih * 128 + d];
  }
  __syncthreads();
  for (int e = threadIdx.x; e < TOPK_ * 128; e += 256) {
    int t = e >> 7, d = e & 127;
    float aq = 0.f, ak = 0.f, av = 0.f;
    for (int k = 0; k < 128; ++k) {
      float sv_ = sel[t][k];
      aq += sv_ * Wsq[(size_t)k * 128 + d];
      ak += sv_ * Wsk[(size_t)k * 128 + d];
      av += sv_ * Wsv[(size_t)k * 128 + d];
    }
    sq[t][d] = aq; sk[t][d] = ak; sv[t][d] = av;
  }
  __syncthreads();
  if (threadIdx.x < 64) {
    int q = threadIdx.x >> 3, k = threadIdx.x & 7;
    float a = 0.f;
    for (int d = 0; d < 128; ++d) a += sq[q][d] * sk[k][d];
    sp[q][k] = a * 0.08838834764831845f;  // 1/sqrt(128)
  }
  __syncthreads();
  if (threadIdx.x < TOPK_) {
    int q = threadIdx.x;
    float mx = -3.4e38f;
    for (int k = 0; k < TOPK_; ++k) mx = fmaxf(mx, sp[q][k]);
    float sm = 0.f, e2[TOPK_];
    for (int k = 0; k < TOPK_; ++k) { e2[k] = __expf(sp[q][k] - mx); sm += e2[k]; }
    for (int k = 0; k < TOPK_; ++k) sp[q][k] = e2[k] / sm;
  }
  __syncthreads();
  for (int e = threadIdx.x; e < TOPK_ * 128; e += 256) {    // so -> reuse sel
    int t = e >> 7, d = e & 127;
    float a = 0.f;
    for (int k = 0; k < TOPK_; ++k) a += sp[t][k] * sv[k][d];
    sel[t][d] = a;
  }
  __syncthreads();
  for (int d = threadIdx.x; d < 128; d += 256) {
    float a = 0.f;
    for (int t = 0; t < TOPK_; ++t) a += sel[t][d];
    mean[d] = a * (1.0f / TOPK_);
  }
  __syncthreads();
  for (int d = threadIdx.x; d < 128; d += 256) {
    float a = 0.f;
    for (int e = 0; e < 128; ++e) a += mean[e] * Wio[(size_t)e * H_ + d];   // only first VD cols
    io_vec[blockIdx.x * 128 + d] = a;
  }
}

// attn_cat[(b*S+s)][h*128+d] = bf16(attn[(b,h,s)][d] + io_vec[b*8 + h/2][d])
__global__ void concat_add_kernel(__bf16* __restrict__ acat, const float* __restrict__ attn,
                                  const float* __restrict__ io_vec, int S) {
  size_t i = (size_t)blockIdx.x * 256 + threadIdx.x;
  if (i >= (size_t)B_ * S * 2048) return;
  int col = (int)(i & 2047); size_t row = i >> 11;
  int h = col >> 7, d = col & 127;
  int s = (int)(row % S); int b = (int)(row / S);
  float v = attn[(((size_t)(b * NH_ + h)) * S + s) * VD_ + d] +
            io_vec[((size_t)(b * IH_ + (h >> 1))) * 128 + d];
  acat[i] = f2bf(v);
}

// ---------------------------------------------------------------------------
// Host launcher
// ---------------------------------------------------------------------------
static inline void* wsalloc(char*& p, size_t bytes) {
  void* r = (void*)p;
  p += (bytes + 255) & ~((size_t)255);
  return r;
}

extern "C" void kernel_launch(void* const* d_in, const int* in_sizes, int n_in,
                              void* d_out, int out_size, void* d_ws, size_t ws_size,
                              hipStream_t stream) {
  (void)in_sizes; (void)n_in; (void)out_size; (void)ws_size;
  const float* x        = (const float*)d_in[0];
  const float* W_cq     = (const float*)d_in[1];
  const float* q_norm_w = (const float*)d_in[2];
  const float* q_norm_b = (const float*)d_in[3];
  const float* W_q      = (const float*)d_in[4];
  const float* W_ckv    = (const float*)d_in[5];
  const float* kv_nw    = (const float*)d_in[6];
  const float* kv_nb    = (const float*)d_in[7];
  const float* W_kv     = (const float*)d_in[8];
  const float* W_o      = (const float*)d_in[9];
  const float* W_ip     = (const float*)d_in[10];
  const float* W_ig     = (const float*)d_in[11];
  const float* W_sq     = (const float*)d_in[12];
  const float* W_sk     = (const float*)d_in[13];
  const float* W_sv     = (const float*)d_in[14];
  const float* W_io     = (const float*)d_in[15];

  const int S = S_;
  const size_t rows = (size_t)B_ * S;           // 4096
  char* p = (char*)d_ws;

  __bf16* x_bf   = (__bf16*)wsalloc(p, rows * H_ * 2);
  __bf16* WcqT   = (__bf16*)wsalloc(p, (size_t)H_ * QLR_ * 2);
  __bf16* WqT    = (__bf16*)wsalloc(p, (size_t)QLR_ * 3072 * 2);
  __bf16* WckvT  = (__bf16*)wsalloc(p, (size_t)H_ * 3072 * 2);
  __bf16* WkvT   = (__bf16*)wsalloc(p, (size_t)128 * 256 * 2);
  __bf16* WoT    = (__bf16*)wsalloc(p, (size_t)2048 * 2048 * 2);
  __bf16* WipT   = (__bf16*)wsalloc(p, (size_t)H_ * 1024 * 2);
  float*  qc     = (float*)wsalloc(p, rows * QLR_ * 4);
  __bf16* qcn    = (__bf16*)wsalloc(p, rows * QLR_ * 2);
  float*  qp     = (float*)wsalloc(p, rows * 3072 * 4);
  float*  kvc    = (float*)wsalloc(p, rows * 3072 * 4);
  __bf16* kvn    = (__bf16*)wsalloc(p, (size_t)B_ * NH_ * S * 128 * 2);
  float*  kvp    = (float*)wsalloc(p, (size_t)B_ * NH_ * S * 256 * 4);
  __bf16* Qm     = (__bf16*)wsalloc(p, (size_t)B_ * NH_ * S * DQK_ * 2);
  __bf16* Km     = (__bf16*)wsalloc(p, (size_t)B_ * NH_ * S * DQK_ * 2);
  __bf16* VTm    = (__bf16*)wsalloc(p, (size_t)B_ * NH_ * VD_ * S * 2);
  float*  attn   = (float*)wsalloc(p, (size_t)B_ * NH_ * S * VD_ * 4);
  float*  ipm    = (float*)wsalloc(p, rows * 1024 * 4);
  float*  gate   = (float*)wsalloc(p, rows * IH_ * 4);
  int*    topidx = (int*)wsalloc(p, (size_t)B_ * IH_ * TOPK_ * 4);
  float*  iovec  = (float*)wsalloc(p, (size_t)B_ * IH_ * 128 * 4);
  __bf16* acat   = (__bf16*)wsalloc(p, rows * 2048 * 2);

  auto gblk = [](size_t n) { return (unsigned)((n + 255) / 256); };

  // converts
  cvt_f32_bf16_kernel<<<gblk(rows * H_), 256, 0, stream>>>(x_bf, x, rows * H_);
  wt_transpose_kernel<<<gblk((size_t)H_ * QLR_),   256, 0, stream>>>(WcqT,  W_cq,  H_,   QLR_);
  wt_transpose_kernel<<<gblk((size_t)QLR_ * 3072), 256, 0, stream>>>(WqT,   W_q,   QLR_, 3072);
  wt_transpose_kernel<<<gblk((size_t)H_ * 3072),   256, 0, stream>>>(WckvT, W_ckv, H_,   3072);
  wt_transpose_kernel<<<gblk((size_t)128 * 256),   256, 0, stream>>>(WkvT,  W_kv,  128,  256);
  wt_transpose_kernel<<<gblk((size_t)2048 * 2048), 256, 0, stream>>>(WoT,   W_o,   2048, 2048);
  wt_transpose_kernel<<<gblk((size_t)H_ * 1024),   256, 0, stream>>>(WipT,  W_ip,  H_,   1024);

  // qc = x @ W_cq ; layernorm
  gemm_bf16_wmma_kernel<<<dim3(QLR_ / TN, (unsigned)(rows / TM)), 256, 0, stream>>>(qc, x_bf, WcqT, (int)rows, QLR_, H_);
  ln_rows_kernel<<<(unsigned)rows, 256, 0, stream>>>(qcn, qc, q_norm_w, q_norm_b, QLR_);
  // qp = qcn @ W_q
  gemm_bf16_wmma_kernel<<<dim3(3072 / TN, (unsigned)(rows / TM)), 256, 0, stream>>>(qp, qcn, WqT, (int)rows, 3072, QLR_);
  // kvc = x @ W_ckv
  gemm_bf16_wmma_kernel<<<dim3(3072 / TN, (unsigned)(rows / TM)), 256, 0, stream>>>(kvc, x_bf, WckvT, (int)rows, 3072, H_);
  // kv layernorm; kvp = kvn @ W_kv
  kv_ln_kernel<<<(unsigned)(B_ * NH_ * S), 128, 0, stream>>>(kvn, kvc, kv_nw, kv_nb, S);
  gemm_bf16_wmma_kernel<<<dim3(256 / TN, (unsigned)((size_t)B_ * NH_ * S / TM)), 256, 0, stream>>>(
      kvp, kvn, WkvT, B_ * NH_ * S, 256, 128);

  // build Q/K (RoPE) and V^T
  build_q_kernel<<<gblk((size_t)B_ * NH_ * S * DQK_), 256, 0, stream>>>(Qm, qp, S);
  build_k_kernel<<<gblk((size_t)B_ * NH_ * S * DQK_), 256, 0, stream>>>(Km, kvp, kvc, S);
  build_vt_kernel<<<gblk((size_t)B_ * NH_ * VD_ * S), 256, 0, stream>>>(VTm, kvp, S);

  // flash attention
  flash_attn_kernel<<<dim3(S / 64, B_ * NH_), 128, 0, stream>>>(attn, Qm, Km, VTm, S);

  // sparse branch
  gemm_bf16_wmma_kernel<<<dim3(1024 / TN, (unsigned)(rows / TM)), 256, 0, stream>>>(ipm, x_bf, WipT, (int)rows, 1024, H_);
  gate_kernel<<<gblk(rows * IH_), 256, 0, stream>>>(gate, x, W_ig, (int)rows);
  topk_kernel<<<B_ * IH_, 256, 0, stream>>>(topidx, gate, S);
  sparse_kernel<<<B_ * IH_, 256, 0, stream>>>(iovec, ipm, topidx, W_sq, W_sk, W_sv, W_io, S);

  // concat + add io ; final projection
  concat_add_kernel<<<gblk(rows * 2048), 256, 0, stream>>>(acat, attn, iovec, S);
  gemm_bf16_wmma_kernel<<<dim3(2048 / TN, (unsigned)(rows / TM)), 256, 0, stream>>>(
      (float*)d_out, acat, WoT, (int)rows, 2048, 2048);
}
